// GeneralizedNonLocal2D_70248485093637
// MI455X (gfx1250) — matrix operand order repaired and checked by reading, not compiled, not run on hardware
//
#include <hip/hip_runtime.h>
#include <math.h>

// ---------------------------------------------------------------------------
// GeneralizedNonLocal2D for MI455X (gfx1250): fused WMMA-bf16 flash attention
// with async global->LDS staging of pre-tiled K/V operands.
// ---------------------------------------------------------------------------
typedef __attribute__((ext_vector_type(16))) __bf16 v16bf;
typedef __attribute__((ext_vector_type(8)))  float  v8f;
typedef __attribute__((ext_vector_type(4)))  int    v4i;

#define NB      2
#define C_DIM   288
#define HH      64
#define WW      64
#define HW      4096
#define M_HEADS 9
#define DHEAD   32
#define KV      1024

#define AS1 __attribute__((address_space(1)))
#define AS3 __attribute__((address_space(3)))

#if defined(__gfx1250__) && __has_builtin(__builtin_amdgcn_global_load_async_to_lds_b128)
#define HAVE_ASYNC 1
#else
#define HAVE_ASYNC 0
#endif

static __device__ __forceinline__ void wait_async0() {
#if HAVE_ASYNC
#if __has_builtin(__builtin_amdgcn_s_wait_asynccnt)
  __builtin_amdgcn_s_wait_asynccnt(0);
#else
  asm volatile("s_wait_asynccnt 0x0" ::: "memory");
#endif
#endif
}

// Linear global -> LDS copy in 16B chunks (async DMA path when available).
static __device__ __forceinline__ void lds_copy16(void* __restrict__ dst,
                                                  const void* __restrict__ src,
                                                  int bytes, int tid, int nthreads) {
#if HAVE_ASYNC
  for (int off = tid * 16; off < bytes; off += nthreads * 16)
    __builtin_amdgcn_global_load_async_to_lds_b128(
        (AS1 v4i*)((const char*)src + off),
        (AS3 v4i*)((char*)dst + off), 0, 0);
#else
  for (int off = tid * 16; off < bytes; off += nthreads * 16)
    *(uint4*)((char*)dst + off) = *(const uint4*)((const char*)src + off);
#endif
}

static __device__ __forceinline__ v8f wmma_bf16(v16bf a, v16bf b, v8f c) {
  // D = A(16x32) * B(32x16) + C, f32 accumulate
  return __builtin_amdgcn_wmma_f32_16x16x32_bf16(false, a, false, b, (short)0, c,
                                                 false, false);
}

// contiguous k order used here for the 16-bit A matrix (16x32), matching B: lane half kh (lane>>4),
// vector element e -> K index along the reduction dim.
static __device__ __forceinline__ int a_pat(int e, int kh) {
  return (kh << 4) | e;
}

static __device__ __forceinline__ v8f v8f_zero() {
  v8f z;
#pragma unroll
  for (int i = 0; i < 8; ++i) z[i] = 0.0f;
  return z;
}

// ---------------------------------------------------------------------------
// 0) convert the four big weight matrices f32 -> bf16
// ---------------------------------------------------------------------------
__global__ void cvt_w_kernel(const float* __restrict__ Wq, const float* __restrict__ Wk,
                             const float* __restrict__ Wv, const float* __restrict__ Wp,
                             __bf16* __restrict__ q, __bf16* __restrict__ k,
                             __bf16* __restrict__ v, __bf16* __restrict__ p) {
  int i = blockIdx.x * blockDim.x + threadIdx.x;
  if (i < C_DIM * C_DIM) {
    q[i] = (__bf16)Wq[i];
    k[i] = (__bf16)Wk[i];
    v[i] = (__bf16)Wv[i];
    p[i] = (__bf16)Wp[i];
  }
}

// ---------------------------------------------------------------------------
// 1) positional factors: pf_x[m][wq][wv][dk], pf_y[m][hq][hu][dk]  (f32)
// ---------------------------------------------------------------------------
__global__ void pos_embed_kernel(const float* __restrict__ Wx, const float* __restrict__ Wy,
                                 float* __restrict__ pfx, float* __restrict__ pfy) {
  const int TOT = M_HEADS * 64 * 32 * DHEAD;   // 589824
  int i = blockIdx.x * blockDim.x + threadIdx.x;
  if (i >= 2 * TOT) return;
  bool isY = (i >= TOT);
  int j = isY ? (i - TOT) : i;
  int dk = j & 31;
  int v  = (j >> 5) & 31;        // kv coordinate (stride 2)
  int a  = (j >> 10) & 63;       // q coordinate (stride 1)
  int m  = j >> 16;
  int o  = m * DHEAD + dk;
  float diff = (float)a - 2.0f * (float)v;     // POS_MAG = 1
  const float* Wm = isY ? Wy : Wx;
  const float c0 = (4.0f / (float)C_DIM) * logf(1000.0f);   // WAVE_LEN = 1000
  float acc = 0.0f;
  for (int f = 0; f < 72; ++f) {
    float dm = diff * __expf(-(float)f * c0);
    acc += __sinf(dm) * Wm[o * 144 + f] + __cosf(dm) * Wm[o * 144 + 72 + f];
  }
  (isY ? pfy : pfx)[j] = acc * 0.70710678f;    // 1/sqrt(2)
}

// ---------------------------------------------------------------------------
// 2) projection GEMM (one wave per 16 output rows):
//    Y[pos][o] = sum_c X[n][c][srcpos(pos)] * W[o][c]
//    MODE 0: bf16 (pos, c) row-major
//    MODE 1: V tiled into WMMA B-operand layout  [kvt][ctile][lane][e]
//    MODE 2: K tiled into WMMA B-operand layout  [m][kvt][sub][lane][e]
// ---------------------------------------------------------------------------
template <bool KVSUB, int MODE>
__global__ void __launch_bounds__(32)
proj_kernel(const float* __restrict__ X, const __bf16* __restrict__ Wb,
            __bf16* __restrict__ Y, int nPos) {
  int nIdx = blockIdx.y;
  int pos0 = blockIdx.x * 16;
  int lane = threadIdx.x;
  const float* Xn = X + (size_t)nIdx * C_DIM * HW;

  __shared__ alignas(32) __bf16 a_t[16][C_DIM];
  for (int c = lane; c < C_DIM; c += 32) {
    const float* src = Xn + (size_t)c * HW;
#pragma unroll 4
    for (int i = 0; i < 16; ++i) {
      int p  = pos0 + i;
      int sp = KVSUB ? (((p >> 5) << 7) + ((p & 31) << 1)) : p;
      a_t[i][c] = (__bf16)src[sp];
    }
  }
  __syncthreads();

  v8f acc[18];
#pragma unroll
  for (int t = 0; t < 18; ++t) acc[t] = v8f_zero();

  int row = lane & 15, kh = lane >> 4;
  for (int kc = 0; kc < 9; ++kc) {
    int c0 = kc * 32;
    v16bf a;
#pragma unroll
    for (int e = 0; e < 16; ++e) a[e] = a_t[row][c0 + a_pat(e, kh)];
#pragma unroll
    for (int t = 0; t < 18; ++t) {
      int o = t * 16 + row;
      v16bf b = *(const v16bf*)(Wb + (size_t)o * C_DIM + c0 + kh * 16);
      acc[t] = wmma_bf16(a, b, acc[t]);
    }
  }

  __bf16* Yn = Y + (size_t)nIdx * nPos * C_DIM;
#pragma unroll
  for (int t = 0; t < 18; ++t)
#pragma unroll
    for (int r = 0; r < 8; ++r) {
      int p = pos0 + r + kh * 8;      // output row (position / kv index)
      int c = t * 16 + row;           // output channel
      __bf16 val = (__bf16)acc[t][r];
      if (MODE == 0) {
        Yn[(size_t)p * C_DIM + c] = val;
      } else if (MODE == 1) {
        // B-operand for P@V: lane' = (kvhalf<<4)|colN, elem = kv&15
        int dest = ((((p >> 5) * 18) + (c >> 4)) * 32 +
                    ((((p >> 4) & 1) << 4) | (c & 15))) * 16 + (p & 15);
        Yn[dest] = val;
      } else {
        // B-operand for Q@K^T: lane' = (dhalf<<4)|kvcol, elem = dk&15
        int mh = c >> 5, dk = c & 31;
        int dest = ((((mh * 32) + (p >> 5)) * 2 + ((p >> 4) & 1)) * 32 +
                    (((dk >> 4) << 4) | (p & 15))) * 16 + (dk & 15);
        Yn[dest] = val;
      }
    }
}

// ---------------------------------------------------------------------------
// 3) context mean over positions + head-mixing softmax weights (tiny)
// ---------------------------------------------------------------------------
__global__ void ctx_mix_kernel(const __bf16* __restrict__ qb, const float* __restrict__ Wc,
                               const float* __restrict__ bc, float* __restrict__ mixw) {
  int nIdx = blockIdx.x;
  int tid  = threadIdx.x;   // 288 threads
  __shared__ float ctx[C_DIM];
  __shared__ float lg[M_HEADS];
  const __bf16* qn = qb + (size_t)nIdx * HW * C_DIM;
  float s = 0.0f;
  for (int p = 0; p < HW; ++p) s += (float)qn[(size_t)p * C_DIM + tid];
  ctx[tid] = s * (1.0f / (float)HW);
  __syncthreads();
  if (tid < M_HEADS) {
    float acc = bc[tid];
    for (int c = 0; c < C_DIM; ++c) acc += ctx[c] * Wc[tid * C_DIM + c];
    lg[tid] = acc;
  }
  __syncthreads();
  if (tid == 0) {
    float mx = lg[0];
    for (int i = 1; i < M_HEADS; ++i) mx = fmaxf(mx, lg[i]);
    float e[M_HEADS], sm = 0.0f;
    for (int i = 0; i < M_HEADS; ++i) { e[i] = __expf(lg[i] - mx); sm += e[i]; }
    for (int i = 0; i < M_HEADS; ++i) mixw[nIdx * M_HEADS + i] = e[i] / sm;
  }
}

// ---------------------------------------------------------------------------
// 4) fused flash attention with positional bias + head mixing.
//    2 waves / block, 16 query rows per wave.  Two passes per head:
//    pass1: lane-local online (max, sumexp), one cross-lane LSE merge/head
//    pass2: normalized mixed P @ V accumulated into 18 f32 WMMA tiles
// ---------------------------------------------------------------------------
__global__ void __launch_bounds__(64)
attn_kernel(const __bf16* __restrict__ qb, const __bf16* __restrict__ kb,
            const __bf16* __restrict__ vb, const float* __restrict__ pfx,
            const float* __restrict__ pfy, const float* __restrict__ mixw,
            __bf16* __restrict__ attn_o) {
  int nIdx = blockIdx.y;
  int pos0 = blockIdx.x * 32;
  int tid  = threadIdx.x;
  int lane = tid & 31, wave = tid >> 5;
  int row  = lane & 15, kh = lane >> 4;

  __shared__ alignas(32) __bf16 q_lds[32][C_DIM];     // 18 KB
  __shared__ alignas(32) __bf16 v_lds[18 * 512];      // 18 KB, B-operand tiles
  __shared__ alignas(32) __bf16 k_lds[2 * 512];       //  2 KB, B-operand tiles
  __shared__ alignas(32) __bf16 p_lds[2 * 512];       //  2 KB, A-operand / wave
  __shared__ float exy[32][64];                       //  8 KB: [row][wv | 32+hu]

  // stage the 32 query rows (bf16, row major) once — linear async copy
  const __bf16* qn = qb + ((size_t)nIdx * HW + pos0) * C_DIM;
  lds_copy16(&q_lds[0][0], qn, 32 * C_DIM * 2, tid, 64);
  wait_async0();
  __syncthreads();

  v8f O[18];
#pragma unroll
  for (int t = 0; t < 18; ++t) O[t] = v8f_zero();

  const __bf16* kn = kb + (size_t)nIdx * KV * C_DIM;  // [m][kvt][sub][lane][e]
  const __bf16* vn = vb + (size_t)nIdx * KV * C_DIM;  // [kvt][ctile][lane][e]

  for (int m = 0; m < M_HEADS; ++m) {
    float wmix = mixw[nIdx * M_HEADS + m];

    // per-row positional bias vectors for this head
    __syncthreads();
    for (int j = tid; j < 32 * 64; j += 64) {
      int rl = j >> 6, col = j & 63;
      int gp = pos0 + rl;
      const float* pf = (col < 32)
          ? pfx + (((size_t)m * 64 + (gp & 63)) * 32 + col) * DHEAD
          : pfy + (((size_t)m * 64 + (gp >> 6)) * 32 + (col - 32)) * DHEAD;
      float acc = 0.0f;
#pragma unroll 8
      for (int dk = 0; dk < DHEAD; ++dk)
        acc += (float)q_lds[rl][m * DHEAD + dk] * pf[dk];
      exy[rl][col] = acc;
    }
    __syncthreads();

    // WMMA A operand for this wave's 16 query rows (reused across KV tiles)
    v16bf aq;
#pragma unroll
    for (int e = 0; e < 16; ++e)
      aq[e] = q_lds[wave * 16 + row][m * DHEAD + a_pat(e, kh)];

    float mr[8], lr[8];
#pragma unroll
    for (int r = 0; r < 8; ++r) { mr[r] = -1e30f; lr[r] = 0.0f; }

    // ---------------- PASS 1: lane-local online max / sum-exp ----------
    for (int kt = 0; kt < 32; ++kt) {
      int kv0 = kt * 32;
      __syncthreads();
      lds_copy16(k_lds, kn + (size_t)(m * 32 + kt) * 1024, 2048, tid, 64);
      wait_async0();
      __syncthreads();
      v16bf b0 = *(const v16bf*)(k_lds + lane * 16);
      v16bf b1 = *(const v16bf*)(k_lds + 512 + lane * 16);
      v8f s0 = wmma_bf16(aq, b0, v8f_zero());
      v8f s1 = wmma_bf16(aq, b1, v8f_zero());
      int kvg0 = kv0 + row, kvg1 = kv0 + 16 + row;
      int hu0 = kvg0 >> 5, wv0 = kvg0 & 31;
      int hu1 = kvg1 >> 5, wv1 = kvg1 & 31;
#pragma unroll
      for (int r = 0; r < 8; ++r) {
        int rl = wave * 16 + r + kh * 8;
        float e0 = s0[r] + exy[rl][wv0] + exy[rl][32 + hu0];
        float e1 = s1[r] + exy[rl][wv1] + exy[rl][32 + hu1];
        float mnew = fmaxf(mr[r], fmaxf(e0, e1));
        lr[r] = lr[r] * __expf(mr[r] - mnew) + __expf(e0 - mnew) + __expf(e1 - mnew);
        mr[r] = mnew;
      }
    }
    // one log-sum-exp merge across the 16-lane row group per head
#pragma unroll
    for (int r = 0; r < 8; ++r) {
#pragma unroll
      for (int st = 1; st <= 8; st <<= 1) {
        float mo = __shfl_xor(mr[r], st);
        float lo = __shfl_xor(lr[r], st);
        float mn = fmaxf(mr[r], mo);
        lr[r] = lr[r] * __expf(mr[r] - mn) + lo * __expf(mo - mn);
        mr[r] = mn;
      }
    }

    float nf[8];
#pragma unroll
    for (int r = 0; r < 8; ++r) nf[r] = wmix / lr[r];

    // ---------------- PASS 2: mixed P @ V accumulation -----------------
    for (int kt = 0; kt < 32; ++kt) {
      int kv0 = kt * 32;
      __syncthreads();
      lds_copy16(k_lds, kn + (size_t)(m * 32 + kt) * 1024, 2048, tid, 64);
      lds_copy16(v_lds, vn + (size_t)kt * 9216, 18432, tid, 64);
      wait_async0();
      if (kt + 1 < 32)
        __builtin_prefetch(vn + (size_t)(kt + 1) * 9216 + tid * 16, 0, 0);
      __syncthreads();

      v16bf b0 = *(const v16bf*)(k_lds + lane * 16);
      v16bf b1 = *(const v16bf*)(k_lds + 512 + lane * 16);
      v8f s0 = wmma_bf16(aq, b0, v8f_zero());
      v8f s1 = wmma_bf16(aq, b1, v8f_zero());
      int kvg0 = kv0 + row, kvg1 = kv0 + 16 + row;
      int hu0 = kvg0 >> 5, wv0 = kvg0 & 31;
      int hu1 = kvg1 >> 5, wv1 = kvg1 & 31;
      __bf16* pw = p_lds + wave * 512;
#pragma unroll
      for (int r = 0; r < 8; ++r) {
        int rl   = wave * 16 + r + kh * 8;
        int rowl = r + kh * 8;
        float e0 = s0[r] + exy[rl][wv0] + exy[rl][32 + hu0];
        float e1 = s1[r] + exy[rl][wv1] + exy[rl][32 + hu1];
        float p0 = __expf(e0 - mr[r]) * nf[r];
        float p1 = __expf(e1 - mr[r]) * nf[r];
        // scatter into A-operand layout (invert a_pat for kvl = row / 16+row)
#pragma unroll
        for (int s = 0; s < 2; ++s) {
          int kvl = s * 16 + row;
          float pv = s ? p1 : p0;
          int khp = kvl >> 4;
          int ep  = kvl & 15;
          pw[(((khp << 4) | rowl) << 4) + ep] = (__bf16)pv;
        }
      }
      // wave-local LDS RAW: DScnt waits inserted by compiler
      v16bf ap = *(const v16bf*)(pw + lane * 16);
#pragma unroll
      for (int t = 0; t < 18; ++t) {
        v16bf bv = *(const v16bf*)(v_lds + t * 512 + lane * 16);
        O[t] = wmma_bf16(ap, bv, O[t]);
      }
    }
  }

  // store mixed attention output, (pos, c) major bf16
  __bf16* on = attn_o + (size_t)nIdx * HW * C_DIM;
#pragma unroll
  for (int t = 0; t < 18; ++t)
#pragma unroll
    for (int r = 0; r < 8; ++r) {
      int p = pos0 + wave * 16 + r + kh * 8;
      on[(size_t)p * C_DIM + t * 16 + row] = (__bf16)O[t][r];
    }
}

// ---------------------------------------------------------------------------
// 5) output projection + bias + gamma residual, writes (n, C, h, w) f32
// ---------------------------------------------------------------------------
__global__ void __launch_bounds__(32)
out_proj_kernel(const __bf16* __restrict__ A, const __bf16* __restrict__ Wb,
                const float* __restrict__ bias, const float* __restrict__ gamma,
                const float* __restrict__ xin, float* __restrict__ out) {
  int nIdx = blockIdx.y;
  int pos0 = blockIdx.x * 16;
  int lane = threadIdx.x;

  __shared__ alignas(32) __bf16 a_t[16][C_DIM];
  const __bf16* An = A + ((size_t)nIdx * HW + pos0) * C_DIM;
  lds_copy16(&a_t[0][0], An, 16 * C_DIM * 2, lane, 32);
  wait_async0();
  __syncthreads();

  v8f acc[18];
#pragma unroll
  for (int t = 0; t < 18; ++t) acc[t] = v8f_zero();

  int row = lane & 15, kh = lane >> 4;
  for (int kc = 0; kc < 9; ++kc) {
    int c0 = kc * 32;
    v16bf a;
#pragma unroll
    for (int e = 0; e < 16; ++e) a[e] = a_t[row][c0 + a_pat(e, kh)];
#pragma unroll
    for (int t = 0; t < 18; ++t) {
      int o = t * 16 + row;
      v16bf b = *(const v16bf*)(Wb + (size_t)o * C_DIM + c0 + kh * 16);
      acc[t] = wmma_bf16(a, b, acc[t]);
    }
  }

  float g = gamma[0];
  const float* xn = xin + (size_t)nIdx * C_DIM * HW;
  float* on = out + (size_t)nIdx * C_DIM * HW;
#pragma unroll
  for (int t = 0; t < 18; ++t)
#pragma unroll
    for (int r = 0; r < 8; ++r) {
      int p = pos0 + r + kh * 8;
      int o = t * 16 + row;
      on[(size_t)o * HW + p] = g * (acc[t][r] + bias[o]) + xn[(size_t)o * HW + p];
    }
}

// ---------------------------------------------------------------------------
extern "C" void kernel_launch(void* const* d_in, const int* in_sizes, int n_in,
                              void* d_out, int out_size, void* d_ws, size_t ws_size,
                              hipStream_t stream) {
  (void)in_sizes; (void)n_in; (void)out_size; (void)ws_size;
  const float* x     = (const float*)d_in[0];
  const float* Wq    = (const float*)d_in[1];
  const float* Wk    = (const float*)d_in[2];
  const float* Wv    = (const float*)d_in[3];
  const float* Wx    = (const float*)d_in[4];
  const float* Wy    = (const float*)d_in[5];
  const float* Wproj = (const float*)d_in[6];
  const float* bproj = (const float*)d_in[7];
  const float* Wc    = (const float*)d_in[8];
  const float* bc    = (const float*)d_in[9];
  const float* gamma = (const float*)d_in[10];
  float* out = (float*)d_out;

  char* ws = (char*)d_ws;
  size_t off = 0;
  auto alloc = [&](size_t bytes) -> void* {
    void* p = ws + off;
    off = (off + bytes + 255) & ~(size_t)255;
    return p;
  };
  __bf16* Wq_b   = (__bf16*)alloc((size_t)C_DIM * C_DIM * 2);
  __bf16* Wk_b   = (__bf16*)alloc((size_t)C_DIM * C_DIM * 2);
  __bf16* Wv_b   = (__bf16*)alloc((size_t)C_DIM * C_DIM * 2);
  __bf16* Wp_b   = (__bf16*)alloc((size_t)C_DIM * C_DIM * 2);
  __bf16* qb     = (__bf16*)alloc((size_t)NB * HW * C_DIM * 2);
  __bf16* kb     = (__bf16*)alloc((size_t)NB * KV * C_DIM * 2);   // K-tiled
  __bf16* vb     = (__bf16*)alloc((size_t)NB * KV * C_DIM * 2);   // V-tiled
  float*  pfx    = (float*)alloc((size_t)M_HEADS * 64 * 32 * DHEAD * 4);
  float*  pfy    = (float*)alloc((size_t)M_HEADS * 64 * 32 * DHEAD * 4);
  float*  mixw   = (float*)alloc((size_t)NB * M_HEADS * 4);
  __bf16* attn_o = (__bf16*)alloc((size_t)NB * HW * C_DIM * 2);

  cvt_w_kernel<<<(C_DIM * C_DIM + 255) / 256, 256, 0, stream>>>(
      Wq, Wk, Wv, Wproj, Wq_b, Wk_b, Wv_b, Wp_b);

  pos_embed_kernel<<<(2 * M_HEADS * 64 * 32 * DHEAD + 255) / 256, 256, 0, stream>>>(
      Wx, Wy, pfx, pfy);

  proj_kernel<false, 0><<<dim3(HW / 16, NB), 32, 0, stream>>>(x, Wq_b, qb, HW);
  proj_kernel<true,  2><<<dim3(KV / 16, NB), 32, 0, stream>>>(x, Wk_b, kb, KV);
  proj_kernel<true,  1><<<dim3(KV / 16, NB), 32, 0, stream>>>(x, Wv_b, vb, KV);

  ctx_mix_kernel<<<NB, C_DIM, 0, stream>>>(qb, Wc, bc, mixw);

  attn_kernel<<<dim3(HW / 32, NB), 64, 0, stream>>>(qb, kb, vb, pfx, pfy, mixw, attn_o);

  out_proj_kernel<<<dim3(HW / 16, NB), 32, 0, stream>>>(attn_o, Wp_b, bproj, gamma, x, out);
}
